// selfAttention_6640019439985
// MI455X (gfx1250) — matrix operand order repaired and checked by reading, not compiled, not run on hardware
//
#include <hip/hip_runtime.h>

typedef __attribute__((ext_vector_type(16))) _Float16 v16h;
typedef __attribute__((ext_vector_type(8)))  _Float16 v8h;
typedef __attribute__((ext_vector_type(8)))  float    v8f;
typedef __attribute__((ext_vector_type(4)))  float    v4f;

#define BATCH 16
#define CCH   256
#define HH    96
#define WW    96
#define HWS   (HH*WW)          // 9216 pixels per (b,c) plane
#define SCALE_ATT 0.0625f      // 1/sqrt(256)

__device__ __forceinline__ v8f wmma_f16(v16h a, v16h b, v8f c) {
  // D = A(16x32 f16) * B(32x16 f16) + C(16x16 f32)
  return __builtin_amdgcn_wmma_f32_16x16x32_f16(false, a, false, b, (short)0, c,
                                                false, false);
}

__device__ __forceinline__ v16h cat8(v8h lo, v8h hi) {
  return __builtin_shufflevector(lo, hi, 0,1,2,3,4,5,6,7,8,9,10,11,12,13,14,15);
}

// A-fragment from an f16 row: e<8 -> k0+half*8+e ; e>=8 -> k0+16+half*8+(e-8)
__device__ __forceinline__ v16h load_a_f16row(const _Float16* __restrict__ row,
                                              int k0, int half) {
  const _Float16* p = row + k0 + half * 8;
  return cat8(*(const v8h*)p, *(const v8h*)(p + 16));
}

// CDNA5 async copy: 16 bytes global -> LDS per lane, tracked by ASYNCcnt.
__device__ __forceinline__ void async_copy_b128(unsigned lds_off,
                                                const void* gaddr) {
  asm volatile("global_load_async_to_lds_b128 %0, %1, off"
               :: "v"(lds_off), "v"(gaddr) : "memory");
}

// ---------------------------------------------------------------------------
// Stage 0: one-shot f32 -> f16 conversion of the three 256x256 weight mats.
// ---------------------------------------------------------------------------
__global__ void __launch_bounds__(256)
convert_w_kernel(const float* __restrict__ Wq, const float* __restrict__ Wk,
                 const float* __restrict__ Wv, _Float16* __restrict__ Wh)
{
  int i = blockIdx.x * 256 + threadIdx.x;        // 0 .. 65535
  Wh[i]              = (_Float16)Wq[i];
  Wh[CCH*CCH + i]    = (_Float16)Wk[i];
  Wh[2*CCH*CCH + i]  = (_Float16)Wv[i];
}

// ---------------------------------------------------------------------------
// Stage 1: Q/K/V = W{q,k,v} @ X + b   as f16 into workspace.
// Q,K layout [b,d,h,w]; V stored transposed [b,d,w,g] for stage-2 B-operand.
// Block: 128 threads (4 waves). Block tile M=64 (out channels) x N=64 (pixels).
// ---------------------------------------------------------------------------
__global__ void __launch_bounds__(128)
qkv_proj_kernel(const float* __restrict__ X,
                const _Float16* __restrict__ Whq, const float* __restrict__ Bq,
                const _Float16* __restrict__ Whk, const float* __restrict__ Bk,
                const _Float16* __restrict__ Whv, const float* __restrict__ Bv,
                _Float16* __restrict__ Qo, _Float16* __restrict__ Ko,
                _Float16* __restrict__ Vt)
{
  const int LDK = 40;                                   // padded k-stride (f16)
  __shared__ __attribute__((aligned(16))) _Float16 XsT[64 * 40]; // [n][k]

  const int b   = blockIdx.z;
  const int n0  = blockIdx.x * 64;
  const int d0  = blockIdx.y * 64;
  const int tid = threadIdx.x;
  const int lane = tid & 31, wave = tid >> 5;
  const int half = lane >> 4, l16 = lane & 15;
  const int dw = d0 + wave * 16;                        // wave's 16 output rows

  const float* Xb = X + (size_t)b * CCH * HWS;
  const _Float16* wqrow = Whq + (size_t)(dw + l16) * CCH;
  const _Float16* wkrow = Whk + (size_t)(dw + l16) * CCH;
  const _Float16* wvrow = Whv + (size_t)(dw + l16) * CCH;

  const v8f vzero = {0.f, 0.f, 0.f, 0.f, 0.f, 0.f, 0.f, 0.f};
  v8f accq[4], acck[4], accv[4];
#pragma unroll
  for (int t = 0; t < 4; ++t) { accq[t] = vzero; acck[t] = vzero; accv[t] = vzero; }

  const int f = tid & 15, krow = tid >> 4;              // staging coords

  for (int k0 = 0; k0 < CCH; k0 += 32) {
    __syncthreads();
    // Stage X[k0:k0+32, n0:n0+64] -> LDS transposed as f16: XsT[n][k]
#pragma unroll
    for (int i = 0; i < 4; ++i) {
      int k = krow + i * 8;
      v4f x = *(const v4f*)(Xb + (size_t)(k0 + k) * HWS + n0 + f * 4);
#pragma unroll
      for (int j = 0; j < 4; ++j)
        XsT[(f * 4 + j) * LDK + k] = (_Float16)x[j];
    }
    __syncthreads();

    v16h aq = load_a_f16row(wqrow, k0, half);
    v16h ak = load_a_f16row(wkrow, k0, half);
    v16h av = load_a_f16row(wvrow, k0, half);
#pragma unroll
    for (int t = 0; t < 4; ++t) {
      // B layout: lane holds column n = t*16+l16, K = half*8 + e (e<8), 16 + half*8 + (e-8) (e>=8)
      const _Float16* p = &XsT[(t * 16 + l16) * LDK + half * 8];
      v16h bb = cat8(*(const v8h*)p, *(const v8h*)(p + 16));
      accq[t] = wmma_f16(aq, bb, accq[t]);
      acck[t] = wmma_f16(ak, bb, acck[t]);
      accv[t] = wmma_f16(av, bb, accv[t]);
    }
  }

  float bq8[8], bk8[8], bv8[8];
#pragma unroll
  for (int r = 0; r < 8; ++r) {
    int d = dw + half * 8 + r;
    bq8[r] = Bq[d]; bk8[r] = Bk[d]; bv8[r] = Bv[d];
  }
#pragma unroll
  for (int t = 0; t < 4; ++t) {
    int n = n0 + t * 16 + l16;
    int g = n / WW, w = n - g * WW;                     // tile never crosses a row
#pragma unroll
    for (int r = 0; r < 8; ++r) {
      int d = dw + half * 8 + r;
      size_t base = ((size_t)b * CCH + d) * HWS;
      Qo[base + n]                  = (_Float16)(accq[t][r] + bq8[r]);
      Ko[base + n]                  = (_Float16)(acck[t][r] + bk8[r]);
      Vt[base + (size_t)w * HH + g] = (_Float16)(accv[t][r] + bv8[r]);
    }
  }
}

// ---------------------------------------------------------------------------
// Stage 2: per (b,c): S = Q K^T * scale -> softmax rows -> P V, + feature_map.
// Block: 192 threads (6 waves); wave owns a 16-row h-strip.
// K and V^T tiles are staged once per block into LDS via async copies.
// ---------------------------------------------------------------------------
__global__ void __launch_bounds__(192)
attention_kernel(const _Float16* __restrict__ Q, const _Float16* __restrict__ K,
                 const _Float16* __restrict__ Vt, const float* __restrict__ fmap,
                 float* __restrict__ out)
{
  __shared__ __attribute__((aligned(16))) float    S[96 * 100];
  __shared__ __attribute__((aligned(16))) _Float16 P[96 * 104];
  __shared__ __attribute__((aligned(16))) _Float16 Ks[96 * 104];
  __shared__ __attribute__((aligned(16))) _Float16 Vs[96 * 104];

  const int tid = threadIdx.x, lane = tid & 31, wave = tid >> 5;  // wave 0..5
  const int half = lane >> 4, l16 = lane & 15;
  const size_t base = (size_t)blockIdx.x * HWS;                   // bc plane
  const _Float16* Qb = Q + base;
  const _Float16* Kb = K + base;
  const _Float16* Vb = Vt + base;
  const int h0 = wave * 16;

  // --- async-stage K (first 6 ops) then V^T (next 6) into LDS ---
  // 96x96 f16 = 1152 16-byte chunks per matrix; 6 chunks per thread.
#pragma unroll
  for (int i = 0; i < 6; ++i) {
    int c = tid + i * 192;
    int row = c / 12, col = (c % 12) * 8;
    unsigned lds = (unsigned)(size_t)(const void*)&Ks[row * 104 + col];
    async_copy_b128(lds, Kb + row * WW + col);
  }
#pragma unroll
  for (int i = 0; i < 6; ++i) {
    int c = tid + i * 192;
    int row = c / 12, col = (c % 12) * 8;
    unsigned lds = (unsigned)(size_t)(const void*)&Vs[row * 104 + col];
    async_copy_b128(lds, Vb + row * HH + col);
  }
  // async loads complete in order: <=6 outstanding means all K chunks landed.
  asm volatile("s_wait_asynccnt 0x6" ::: "memory");
  __syncthreads();

  const v8f vzero = {0.f, 0.f, 0.f, 0.f, 0.f, 0.f, 0.f, 0.f};

  // --- GEMM1: S[h0:h0+16, :] = Q strip @ K^T ---
  v8f acc[6];
#pragma unroll
  for (int t = 0; t < 6; ++t) acc[t] = vzero;
#pragma unroll
  for (int k0 = 0; k0 < WW; k0 += 32) {
    const _Float16* pa = Qb + (size_t)(h0 + l16) * WW + k0 + half * 8;
    v16h a = cat8(*(const v8h*)pa, *(const v8h*)(pa + 16));
#pragma unroll
    for (int t = 0; t < 6; ++t) {
      const _Float16* pb = &Ks[(t * 16 + l16) * 104 + k0 + half * 8];
      v16h bb = cat8(*(const v8h*)pb, *(const v8h*)(pb + 16));
      acc[t] = wmma_f16(a, bb, acc[t]);
    }
  }
#pragma unroll
  for (int t = 0; t < 6; ++t)
#pragma unroll
    for (int r = 0; r < 8; ++r)
      S[(h0 + half * 8 + r) * 100 + t * 16 + l16] = acc[t][r] * SCALE_ATT;

  // V^T staging must be fully landed (own wave) before the barrier below.
  asm volatile("s_wait_asynccnt 0x0" ::: "memory");
  __syncthreads();

  // --- row softmax (f32), P = softmax(S) as f16 in A-friendly layout ---
  if (tid < 96) {
    float* row = &S[tid * 100];
    float m = -3.402823466e38f;
    for (int j = 0; j < 96; ++j) m = fmaxf(m, row[j]);
    float s = 0.f;
    for (int j = 0; j < 96; ++j) { float e = __expf(row[j] - m); row[j] = e; s += e; }
    float inv = 1.0f / s;
    _Float16* prow = &P[tid * 104];
    for (int j = 0; j < 96; ++j) prow[j] = (_Float16)(row[j] * inv);
  }
  __syncthreads();

  // --- GEMM2: att[h0:h0+16, :] = P strip @ V  (B operand from LDS V^T) ---
  v8f acc2[6];
#pragma unroll
  for (int t = 0; t < 6; ++t) acc2[t] = vzero;
#pragma unroll
  for (int k0 = 0; k0 < HH; k0 += 32) {
    const _Float16* pa = &P[(h0 + l16) * 104 + k0 + half * 8];
    v16h a = cat8(*(const v8h*)pa, *(const v8h*)(pa + 16));
#pragma unroll
    for (int t = 0; t < 6; ++t) {
      const _Float16* pb = &Vs[(t * 16 + l16) * 104 + k0 + half * 8];
      v16h bb = cat8(*(const v8h*)pb, *(const v8h*)(pb + 16));
      acc2[t] = wmma_f16(a, bb, acc2[t]);
    }
  }
#pragma unroll
  for (int t = 0; t < 6; ++t) {
#pragma unroll
    for (int r = 0; r < 8; ++r) {
      int h = h0 + half * 8 + r;
      int w = t * 16 + l16;
      size_t idx = base + (size_t)h * WW + w;
      out[idx] = fmap[idx] + acc2[t][r];
    }
  }
}

// ---------------------------------------------------------------------------
extern "C" void kernel_launch(void* const* d_in, const int* in_sizes, int n_in,
                              void* d_out, int out_size, void* d_ws, size_t ws_size,
                              hipStream_t stream)
{
  (void)in_sizes; (void)n_in; (void)out_size; (void)ws_size;
  const float* feature = (const float*)d_in[0];
  const float* fmap    = (const float*)d_in[1];
  const float* Wq = (const float*)d_in[2];
  const float* Bq = (const float*)d_in[3];
  const float* Wk = (const float*)d_in[4];
  const float* Bk = (const float*)d_in[5];
  const float* Wv = (const float*)d_in[6];
  const float* Bv = (const float*)d_in[7];
  float* out = (float*)d_out;

  const size_t elems = (size_t)BATCH * CCH * HWS;      // 37,748,736
  _Float16* Qo = (_Float16*)d_ws;
  _Float16* Ko = (_Float16*)((char*)d_ws + 2 * elems);
  _Float16* Vt = (_Float16*)((char*)d_ws + 4 * elems);
  _Float16* Wh = (_Float16*)((char*)d_ws + 6 * elems); // 3 x 256x256 f16

  convert_w_kernel<<<dim3(CCH * CCH / 256), dim3(256), 0, stream>>>(Wq, Wk, Wv, Wh);

  dim3 g1(HWS / 64, CCH / 64, BATCH);                  // (144, 4, 16)
  qkv_proj_kernel<<<g1, dim3(128), 0, stream>>>(feature,
                                                Wh, Bq,
                                                Wh + CCH * CCH, Bk,
                                                Wh + 2 * CCH * CCH, Bv,
                                                Qo, Ko, Vt);
  attention_kernel<<<dim3(BATCH * CCH), dim3(192), 0, stream>>>(Qo, Ko, Vt,
                                                                fmap, out);
}
